// DDiTBlockWithMask_41489384079541
// MI455X (gfx1250) — matrix operand order repaired and checked by reading, not compiled, not run on hardware
//
#include <hip/hip_runtime.h>
#include <hip/hip_bf16.h>

#define DEV __device__ __forceinline__

typedef __bf16 bf16_t;
typedef __attribute__((ext_vector_type(16))) __bf16 v16bf;
typedef __attribute__((ext_vector_type(8)))  __bf16 v8bf;
typedef __attribute__((ext_vector_type(8)))  float   v8f;

static constexpr int BATCH = 2;
static constexpr int SEQ   = 4096;
static constexpr int DIM   = 768;
static constexpr int NH    = 12;
static constexpr int HD    = 64;
static constexpr int QKV_N = 3 * DIM;   // 2304
static constexpr int MLP   = 4 * DIM;   // 3072
static constexpr int MOD6  = 6 * DIM;   // 4608
static constexpr int MROWS = BATCH * SEQ; // 8192

// ---------- scalar bf16 helpers (bit-exact RNE, no reliance on fptrunc) ----
DEV bf16_t f2b(float f) {
    unsigned u = __builtin_bit_cast(unsigned, f);
    unsigned r = (u + 0x7FFFu + ((u >> 16) & 1u)) >> 16;
    return __builtin_bit_cast(bf16_t, (unsigned short)r);
}

// ---------- WMMA wrapper ---------------------------------------------------
DEV v8f wmma_bf16(v16bf a, v16bf b, v8f c) {
    return __builtin_amdgcn_wmma_f32_16x16x32_bf16(
        /*neg_a=*/false, a, /*neg_b=*/false, b,
        /*c_mod=*/(short)0, c, /*reuse_a=*/false, /*reuse_b=*/false);
}

// ---------- CDNA5 async global->LDS copy (ASYNCcnt-tracked) ----------------
// VDST = per-lane LDS byte offset, VADDR = per-lane 64-bit global address.
DEV void async_copy_b128(const bf16_t* g, bf16_t* l) {
    unsigned lds_off = (unsigned)(unsigned long long)(uintptr_t)l; // low 32 bits = LDS offset
    unsigned long long ga = (unsigned long long)(uintptr_t)g;
    asm volatile("global_load_async_to_lds_b128 %0, %1, off"
                 :: "v"(lds_off), "v"(ga)
                 : "memory");
}
DEV void wait_async_le2() { asm volatile("s_wait_asynccnt 0x2" ::: "memory"); }
DEV void wait_async_0()   { asm volatile("s_wait_asynccnt 0x0" ::: "memory"); }

// A fragment: 16x32 (MxK) bf16 from row-major [M][K] source.
// Lane layout (ISA 7.12.2): lanes<16 hold K 0..7 & 16..23, lanes>=16 hold 8..15 & 24..31.
DEV v16bf load_a_frag(const bf16_t* p0, int ld) {
    const int lane = threadIdx.x & 31;
    const bf16_t* p = p0 + (lane & 15) * ld + ((lane >> 4) << 3);
    v8bf lo = *reinterpret_cast<const v8bf*>(p);
    v8bf hi = *reinterpret_cast<const v8bf*>(p + 16);
    v16bf v;
#pragma unroll
    for (int t = 0; t < 8; ++t) { v[t] = lo[t]; v[t + 8] = hi[t]; }
    return v;
}

// B fragment: 32x16 (KxN) bf16, source given row-major [N][K] (i.e. B^T).
// Lanes<16 hold K 0..7 & 16..23, lanes>=16 hold K 8..15 & 24..31 for column N = lane&15.
DEV v16bf load_b_frag(const bf16_t* p0, int ld) {
    const int lane = threadIdx.x & 31;
    const bf16_t* p = p0 + (lane & 15) * ld + ((lane >> 4) << 3);
    v8bf lo = *reinterpret_cast<const v8bf*>(p);
    v8bf hi = *reinterpret_cast<const v8bf*>(p + 16);
    v16bf v;
#pragma unroll
    for (int t = 0; t < 8; ++t) { v[t] = lo[t]; v[t + 8] = hi[t]; }
    return v;
}

// ---------- small helper kernels -------------------------------------------
__global__ __launch_bounds__(256) void to_bf16_kernel(const float* __restrict__ in,
                                                      bf16_t* __restrict__ out, int n) {
    int i = blockIdx.x * 256 + threadIdx.x;
    if (i < n) out[i] = f2b(in[i]);
}

// mod[b][j] = bada[j] + sum_k c[b][k] * Wada[j][k]
__global__ __launch_bounds__(256) void ada_mod_kernel(const float* __restrict__ c,
                                                      const float* __restrict__ Wada,
                                                      const float* __restrict__ bada,
                                                      float* __restrict__ modv) {
    int o = blockIdx.x * 256 + threadIdx.x;
    if (o >= BATCH * MOD6) return;
    int bi = o / MOD6, j = o % MOD6;
    const float* cr = c + bi * DIM;
    const float* wr = Wada + (size_t)j * DIM;
    float s = bada[j];
    for (int k = 0; k < DIM; ++k) s += cr[k] * wr[k];
    modv[o] = s;
}

// y = LN(x) * w * (1 + scale) + shift  ->  bf16
__global__ __launch_bounds__(256) void ln_mod_kernel(const float* __restrict__ X,
                                                     const float* __restrict__ w,
                                                     const float* __restrict__ modv,
                                                     int shift_chunk, int scale_chunk,
                                                     bf16_t* __restrict__ out) {
    const int row = blockIdx.x;
    const int bb  = row >> 12;          // row / SEQ
    const int tid = threadIdx.x;
    const float* xr = X + (size_t)row * DIM;
    float v[3]; float s = 0.f, sq = 0.f;
#pragma unroll
    for (int t = 0; t < 3; ++t) {
        float x = xr[tid + 256 * t];
        v[t] = x; s += x; sq += x * x;
    }
    __shared__ float ss[256], s2[256];
    ss[tid] = s; s2[tid] = sq;
    __syncthreads();
    for (int st = 128; st > 0; st >>= 1) {
        if (tid < st) { ss[tid] += ss[tid + st]; s2[tid] += s2[tid + st]; }
        __syncthreads();
    }
    const float inv = 1.0f / (float)DIM;
    float mu   = ss[0] * inv;
    float var  = s2[0] * inv - mu * mu;
    float rstd = rsqrtf(var + 1e-5f);
#pragma unroll
    for (int t = 0; t < 3; ++t) {
        int n = tid + 256 * t;
        float y = (v[t] - mu) * rstd * w[n];
        y = y * (1.f + modv[bb * MOD6 + scale_chunk * DIM + n])
              + modv[bb * MOD6 + shift_chunk * DIM + n];
        out[(size_t)row * DIM + n] = f2b(y);
    }
}

// RoPE on q/k/v + split into Q[b,h,s,hd], K[b,h,s,hd], V^T[b,h,hd,s] (bf16)
__global__ __launch_bounds__(256) void rope_split_kernel(const float* __restrict__ qkv,
                                                         const float* __restrict__ cosb,
                                                         const float* __restrict__ sinb,
                                                         bf16_t* __restrict__ qb,
                                                         bf16_t* __restrict__ kb,
                                                         bf16_t* __restrict__ vtb) {
    int idx = blockIdx.x * 256 + threadIdx.x;
    if (idx >= MROWS * QKV_N) return;
    int col  = idx % QKV_N;
    int m    = idx / QKV_N;
    int bi   = m >> 12;
    int srow = m & (SEQ - 1);
    int t    = col / DIM;
    int rem  = col % DIM;
    int hh   = rem >> 6, d = rem & 63;
    float x     = qkv[idx];
    float other = (d < 32) ? -qkv[idx + 32] : qkv[idx - 32];
    int sd = srow * HD + d;
    float y = x * cosb[sd] + other * sinb[sd];
    bf16_t yb = f2b(y);
    int bhs = (bi * NH + hh) * SEQ + srow;
    if (t == 0)      qb[(size_t)bhs * HD + d] = yb;
    else if (t == 1) kb[(size_t)bhs * HD + d] = yb;
    else             vtb[(size_t)((bi * NH + hh) * HD + d) * SEQ + srow] = yb;
}

// ---------- WMMA GEMM: 4 waves/WG, 64x64 per wave, async-LDS B staging -----
// out[m][n] = sum_k A[m][k] * Bm[n][k]   (Bm row-major [N][K])
// mode 0: Cf = acc
// mode 1: Cf = res + mod[chunk] * acc
// mode 2: Cb = bf16(gelu_tanh(acc + bias))
// mode 3: Cf = res + mod[chunk] * (acc + bias)
__global__ __launch_bounds__(128) void gemm64_kernel(const bf16_t* __restrict__ A,
                                                     const bf16_t* __restrict__ Bm,
                                                     int N, int K, int mode,
                                                     const float* __restrict__ res,
                                                     const float* __restrict__ modv, int chunk,
                                                     const float* __restrict__ bias,
                                                     float* __restrict__ Cf,
                                                     bf16_t* __restrict__ Cb) {
    const int nbase = blockIdx.x * 64;
    const int wv    = threadIdx.x >> 5;
    const int mbase = blockIdx.y * 256 + wv * 64;
    const int lane  = threadIdx.x & 31;
    const int half  = lane >> 4, nl = lane & 15;
    const int tid   = threadIdx.x;

    // double-buffered B tile: 64 rows(N) x 32 K bf16 per buffer (4 KB each)
    __shared__ __align__(16) bf16_t btile[2][64 * 32];

    // each of 128 threads copies two 16-byte chunks per k-step
    const int c0 = tid * 2;
    const int bn0 = c0 >> 2, bk0 = (c0 & 3) * 8;
    const int c1 = c0 + 1;
    const int bn1 = c1 >> 2, bk1 = (c1 & 3) * 8;

    const v8f zero = {0.f, 0.f, 0.f, 0.f, 0.f, 0.f, 0.f, 0.f};
    v8f acc[4][4];
#pragma unroll
    for (int i = 0; i < 4; ++i)
#pragma unroll
        for (int j = 0; j < 4; ++j) acc[i][j] = zero;

    auto issue = [&](int buf, int kbs) {
        async_copy_b128(Bm + (size_t)(nbase + bn0) * K + kbs + bk0,
                        &btile[buf][bn0 * 32 + bk0]);
        async_copy_b128(Bm + (size_t)(nbase + bn1) * K + kbs + bk1,
                        &btile[buf][bn1 * 32 + bk1]);
    };

    const int nsteps = K >> 5;
    issue(0, 0);
    for (int s = 0; s < nsteps; ++s) {
        const int kbs = s << 5;
        const int cur = s & 1;
        if (s + 1 < nsteps) {
            issue(cur ^ 1, kbs + 32);
            wait_async_le2();       // current buffer's copies complete (in-order)
        } else {
            wait_async_0();
        }
        __syncthreads();            // make staged B visible to all 4 waves

        v16bf af[4], bfm[4];
#pragma unroll
        for (int i = 0; i < 4; ++i)
            af[i] = load_a_frag(A + (size_t)(mbase + 16 * i) * K + kbs, K);
#pragma unroll
        for (int j = 0; j < 4; ++j)
            bfm[j] = load_b_frag(btile[cur] + 16 * j * 32, 32);
#pragma unroll
        for (int i = 0; i < 4; ++i)
#pragma unroll
            for (int j = 0; j < 4; ++j)
                acc[i][j] = wmma_bf16(af[i], bfm[j], acc[i][j]);

        __syncthreads();            // protect buffer reuse two steps ahead
    }

#pragma unroll
    for (int i = 0; i < 4; ++i)
#pragma unroll
        for (int j = 0; j < 4; ++j)
#pragma unroll
            for (int r = 0; r < 8; ++r) {
                int m = mbase + 16 * i + r + 8 * half;
                int n = nbase + 16 * j + nl;
                size_t off = (size_t)m * N + n;
                float a = acc[i][j][r];
                int bb = m >> 12;
                if (mode == 0) {
                    Cf[off] = a;
                } else if (mode == 1) {
                    Cf[off] = res[off] + modv[bb * MOD6 + chunk * DIM + n] * a;
                } else if (mode == 2) {
                    float t = a + bias[n];
                    float g = 0.5f * t *
                        (1.f + tanhf(0.7978845608028654f * (t + 0.044715f * t * t * t)));
                    Cb[off] = f2b(g);
                } else {
                    Cf[off] = res[off] +
                              modv[bb * MOD6 + chunk * DIM + n] * (a + bias[n]);
                }
            }
}

// ---------- flash attention: 1 wave per 16-row query tile ------------------
__global__ __launch_bounds__(32) void flash_attn_kernel(const bf16_t* __restrict__ Qb,
                                                        const bf16_t* __restrict__ Kbm,
                                                        const bf16_t* __restrict__ Vtb,
                                                        const int* __restrict__ amask,
                                                        bf16_t* __restrict__ Ob) {
    const int lane = threadIdx.x;
    const int half = lane >> 4, nl = lane & 15;
    const int bh = blockIdx.y, bi = bh / NH, hh = bh % NH;
    const int qbase = blockIdx.x * 16;
    const bf16_t* Q  = Qb  + (size_t)bh * SEQ * HD;
    const bf16_t* Kp = Kbm + (size_t)bh * SEQ * HD;
    const bf16_t* Vt = Vtb + (size_t)bh * HD * SEQ;
    const int* mrow = amask + bi * SEQ;
    __shared__ __align__(16) bf16_t plds[16 * 64];

    v16bf qf0 = load_a_frag(Q + (size_t)qbase * HD + 0, HD);
    v16bf qf1 = load_a_frag(Q + (size_t)qbase * HD + 32, HD);

    float mq[8];
#pragma unroll
    for (int r = 0; r < 8; ++r) mq[r] = (float)mrow[qbase + r + 8 * half];

    const v8f zero = {0.f, 0.f, 0.f, 0.f, 0.f, 0.f, 0.f, 0.f};
    v8f oacc[4];
    float m_run[8], l_run[8];
#pragma unroll
    for (int jd = 0; jd < 4; ++jd) oacc[jd] = zero;
#pragma unroll
    for (int r = 0; r < 8; ++r) { m_run[r] = -3.0e38f; l_run[r] = 0.f; }

    for (int kbb = 0; kbb < SEQ; kbb += 64) {
        float mk[4];
#pragma unroll
        for (int j = 0; j < 4; ++j) mk[j] = (float)mrow[kbb + 16 * j + nl];

        // S = Q @ K^T * 0.125 + mask bias
        v8f sacc[4];
#pragma unroll
        for (int j = 0; j < 4; ++j) {
            v16bf kf0 = load_b_frag(Kp + (size_t)(kbb + 16 * j) * HD + 0, HD);
            v16bf kf1 = load_b_frag(Kp + (size_t)(kbb + 16 * j) * HD + 32, HD);
            v8f a = zero;
            a = wmma_bf16(qf0, kf0, a);
            a = wmma_bf16(qf1, kf1, a);
#pragma unroll
            for (int r = 0; r < 8; ++r) {
                float biasv = (mq[r] * mk[j] - 1.f) * 1.0e9f;
                a[r] = a[r] * 0.125f + biasv;
            }
            sacc[j] = a;
        }

        // online softmax (reductions stay inside 16-lane halves: xor 1,2,4,8)
        float mnew[8], fscale[8];
#pragma unroll
        for (int r = 0; r < 8; ++r) {
            float v = fmaxf(fmaxf(sacc[0][r], sacc[1][r]),
                            fmaxf(sacc[2][r], sacc[3][r]));
            v = fmaxf(v, __shfl_xor(v, 1, 32));
            v = fmaxf(v, __shfl_xor(v, 2, 32));
            v = fmaxf(v, __shfl_xor(v, 4, 32));
            v = fmaxf(v, __shfl_xor(v, 8, 32));
            float mn = fmaxf(m_run[r], v);
            fscale[r] = __expf(m_run[r] - mn);
            m_run[r] = mn; mnew[r] = mn;
        }
        float rsum[8];
#pragma unroll
        for (int r = 0; r < 8; ++r) rsum[r] = 0.f;
#pragma unroll
        for (int j = 0; j < 4; ++j)
#pragma unroll
            for (int r = 0; r < 8; ++r) {
                float p = __expf(sacc[j][r] - mnew[r]);
                sacc[j][r] = p; rsum[r] += p;
            }
#pragma unroll
        for (int r = 0; r < 8; ++r) {
            float v = rsum[r];
            v += __shfl_xor(v, 1, 32);
            v += __shfl_xor(v, 2, 32);
            v += __shfl_xor(v, 4, 32);
            v += __shfl_xor(v, 8, 32);
            l_run[r] = l_run[r] * fscale[r] + v;
        }
#pragma unroll
        for (int jd = 0; jd < 4; ++jd)
#pragma unroll
            for (int r = 0; r < 8; ++r) oacc[jd][r] *= fscale[r];

        // transpose P (C-layout -> A-fragment layout) through LDS
        __syncthreads();
#pragma unroll
        for (int j = 0; j < 4; ++j)
#pragma unroll
            for (int r = 0; r < 8; ++r)
                plds[(r + 8 * half) * 64 + 16 * j + nl] = f2b(sacc[j][r]);
        __syncthreads();

        v16bf pf0 = load_a_frag(plds + 0, 64);
        v16bf pf1 = load_a_frag(plds + 32, 64);
#pragma unroll
        for (int jd = 0; jd < 4; ++jd) {
            v16bf vf0 = load_b_frag(Vt + (size_t)(16 * jd) * SEQ + kbb + 0, SEQ);
            v16bf vf1 = load_b_frag(Vt + (size_t)(16 * jd) * SEQ + kbb + 32, SEQ);
            oacc[jd] = wmma_bf16(pf0, vf0, oacc[jd]);
            oacc[jd] = wmma_bf16(pf1, vf1, oacc[jd]);
        }
    }

    // O /= l, write to [b, s, h*hd] bf16 (A-operand for out-projection GEMM)
#pragma unroll
    for (int jd = 0; jd < 4; ++jd)
#pragma unroll
        for (int r = 0; r < 8; ++r) {
            int row = qbase + r + 8 * half;
            int d = 16 * jd + nl;
            Ob[(size_t)(bi * SEQ + row) * DIM + hh * HD + d] =
                f2b(oacc[jd][r] / l_run[r]);
        }
}

// ---------------------------------------------------------------------------
extern "C" void kernel_launch(void* const* d_in, const int* in_sizes, int n_in,
                              void* d_out, int out_size, void* d_ws, size_t ws_size,
                              hipStream_t stream) {
    (void)in_sizes; (void)n_in; (void)out_size; (void)ws_size;
    const float* x    = (const float*)d_in[0];
    const float* cosb = (const float*)d_in[1];
    const float* sinb = (const float*)d_in[2];
    const float* c    = (const float*)d_in[3];
    const int*   am   = (const int*)d_in[4];
    const float* ln1w = (const float*)d_in[5];
    const float* Wqkv = (const float*)d_in[6];
    const float* Wout = (const float*)d_in[7];
    const float* ln2w = (const float*)d_in[8];
    const float* W1   = (const float*)d_in[9];
    const float* b1   = (const float*)d_in[10];
    const float* W2   = (const float*)d_in[11];
    const float* b2   = (const float*)d_in[12];
    const float* Wada = (const float*)d_in[13];
    const float* bada = (const float*)d_in[14];

    char* wsp = (char*)d_ws;
    auto alloc = [&](size_t bytes) -> char* {
        char* p = wsp;
        wsp += (bytes + 255) & ~(size_t)255;
        return p;
    };
    float*  modv  = (float*)alloc((size_t)BATCH * MOD6 * 4);
    bf16_t* hdnb  = (bf16_t*)alloc((size_t)MROWS * DIM * 2);
    bf16_t* wqkvb = (bf16_t*)alloc((size_t)QKV_N * DIM * 2);
    bf16_t* woutb = (bf16_t*)alloc((size_t)DIM * DIM * 2);
    bf16_t* w1b   = (bf16_t*)alloc((size_t)MLP * DIM * 2);
    bf16_t* w2b   = (bf16_t*)alloc((size_t)DIM * MLP * 2);
    float*  qkvf  = (float*)alloc((size_t)MROWS * QKV_N * 4);
    bf16_t* qb    = (bf16_t*)alloc((size_t)MROWS * DIM * 2);
    bf16_t* kbm   = (bf16_t*)alloc((size_t)MROWS * DIM * 2);
    bf16_t* vtb   = (bf16_t*)alloc((size_t)MROWS * DIM * 2);
    bf16_t* ob    = (bf16_t*)alloc((size_t)MROWS * DIM * 2);
    float*  x2f   = (float*)alloc((size_t)MROWS * DIM * 4);
    bf16_t* h2b   = (bf16_t*)alloc((size_t)MROWS * DIM * 2);
    bf16_t* gelb  = (bf16_t*)alloc((size_t)MROWS * MLP * 2);

    // weight conversions to bf16
    {
        int n = QKV_N * DIM;
        to_bf16_kernel<<<(n + 255) / 256, 256, 0, stream>>>(Wqkv, wqkvb, n);
        n = DIM * DIM;
        to_bf16_kernel<<<(n + 255) / 256, 256, 0, stream>>>(Wout, woutb, n);
        n = MLP * DIM;
        to_bf16_kernel<<<(n + 255) / 256, 256, 0, stream>>>(W1, w1b, n);
        n = DIM * MLP;
        to_bf16_kernel<<<(n + 255) / 256, 256, 0, stream>>>(W2, w2b, n);
    }

    // adaLN modulation
    ada_mod_kernel<<<(BATCH * MOD6 + 255) / 256, 256, 0, stream>>>(c, Wada, bada, modv);

    // LN1 + modulate (shift=chunk0, scale=chunk1) -> bf16
    ln_mod_kernel<<<MROWS, 256, 0, stream>>>(x, ln1w, modv, 0, 1, hdnb);

    // QKV GEMM: [8192,768] x [2304,768]^T -> f32 [8192,2304]
    gemm64_kernel<<<dim3(QKV_N / 64, MROWS / 256), 128, 0, stream>>>(
        hdnb, wqkvb, QKV_N, DIM, 0, nullptr, nullptr, 0, nullptr, qkvf, nullptr);

    // RoPE + split q/k/v^T (bf16)
    {
        int n = MROWS * QKV_N;
        rope_split_kernel<<<(n + 255) / 256, 256, 0, stream>>>(qkvf, cosb, sinb,
                                                               qb, kbm, vtb);
    }

    // flash attention -> o bf16 [b,s,dim]
    flash_attn_kernel<<<dim3(SEQ / 16, BATCH * NH), 32, 0, stream>>>(qb, kbm, vtb,
                                                                     am, ob);

    // out-proj + gated residual: x2 = x + gate_msa(chunk2) * (o @ Wout^T)
    gemm64_kernel<<<dim3(DIM / 64, MROWS / 256), 128, 0, stream>>>(
        ob, woutb, DIM, DIM, 1, x, modv, 2, nullptr, x2f, nullptr);

    // LN2 + modulate (shift=chunk3, scale=chunk4) -> bf16
    ln_mod_kernel<<<MROWS, 256, 0, stream>>>(x2f, ln2w, modv, 3, 4, h2b);

    // MLP1: gelu(h2 @ W1^T + b1) -> bf16 [8192,3072]
    gemm64_kernel<<<dim3(MLP / 64, MROWS / 256), 128, 0, stream>>>(
        h2b, w1b, MLP, DIM, 2, nullptr, nullptr, 0, b1, nullptr, gelb);

    // MLP2 + gated residual: out = x2 + gate_mlp(chunk5) * (g @ W2^T + b2)
    gemm64_kernel<<<dim3(DIM / 64, MROWS / 256), 128, 0, stream>>>(
        gelb, w2b, DIM, MLP, 3, x2f, modv, 5, b2, (float*)d_out, nullptr);
}
